// GCN_21758304322054
// MI455X (gfx1250) — matrix operand order repaired and checked by reading, not compiled, not run on hardware
//
#include <hip/hip_runtime.h>

// ---------------------------------------------------------------------------
// GCN (2-layer) for MI455X / gfx1250.
// Dense transforms via v_wmma_f32_16x16x32_bf16 (f32 accumulate), one wave
// per 16-row output strip (A fragment reused across all column tiles,
// independent accumulators let B-loads overlap WMMA issue).
// Sparse aggregation: wave-per-edge coalesced gather + f32 atomic scatter;
// whole 51MB feature matrix lives in the 192MB L2, so scatter/gather is
// L2-resident rather than HBM-bound.
// ---------------------------------------------------------------------------

typedef __attribute__((ext_vector_type(16))) __bf16 v16bf;
typedef __attribute__((ext_vector_type(8)))  __bf16 v8bf;
typedef __attribute__((ext_vector_type(8)))  float  v8f;

#define NN 100000   // nodes (multiple of 16: 6250 row tiles)

// ---------------- elementwise / degree kernels ----------------

__global__ __launch_bounds__(256) void k_fill_ones(float* p, int n) {
    int i = blockIdx.x * 256 + threadIdx.x;
    if (i < n) p[i] = 1.0f;                    // self-loop contributes 1 to degree
}

__global__ __launch_bounds__(256) void k_count_deg(const int* dst, float* deg, int E) {
    int e = blockIdx.x * 256 + threadIdx.x;
    if (e < E) atomicAdd(&deg[dst[e]], 1.0f);
}

__global__ __launch_bounds__(256) void k_rsqrt(float* p, int n) {
    int i = blockIdx.x * 256 + threadIdx.x;
    if (i < n) p[i] = rsqrtf(p[i]);            // deg >= 1 always (self-loop)
}

// W[k][m] (f32, row-major KxM)  ->  Wt[m][k] (bf16) so B-fragments are contiguous
__global__ __launch_bounds__(256) void k_w_transpose_bf16(const float* W, __bf16* Wt,
                                                          int K, int M) {
    int i = blockIdx.x * 256 + threadIdx.x;
    if (i < K * M) {
        int k = i / M, m = i % M;
        Wt[m * K + k] = (__bf16)W[i];
    }
}

// agg[i,c] = dinv[i]^2 * h[i,c] (+ bias[c])  -- self-loop message + optional bias
__global__ __launch_bounds__(256) void k_selfloop_init(const float* h, const float* dinv,
                                                       const float* bias, float* out,
                                                       int n, int D) {
    int i = blockIdx.x * 256 + threadIdx.x;
    if (i < n * D) {
        int node = i / D, c = i % D;
        float s = dinv[node];
        float v = s * s * h[i];
        if (bias) v += bias[c];
        out[i] = v;
    }
}

// hbf[i,c] = bf16(relu(agg[i,c] + b[c]))
__global__ __launch_bounds__(256) void k_bias_relu_bf16(const float* agg, const float* b,
                                                        __bf16* out, int n, int D) {
    int i = blockIdx.x * 256 + threadIdx.x;
    if (i < n * D) {
        float v = agg[i] + b[i % D];
        out[i] = (__bf16)fmaxf(v, 0.0f);
    }
}

// ---------------- edge aggregation: one wave per edge ----------------
// lane handles PER = D/32 consecutive channels; gather is one coalesced row.

template <int PER>
__global__ __launch_bounds__(256) void k_scatter(const int* __restrict__ src,
                                                 const int* __restrict__ dst,
                                                 const float* __restrict__ dinv,
                                                 const float* __restrict__ h,
                                                 float* __restrict__ agg, int E) {
    const int D = PER * 32;
    unsigned t = blockIdx.x * 256u + threadIdx.x;
    int e = (int)(t >> 5);
    int lane = (int)(t & 31u);
    if (e >= E) return;

    int s = 0, d = 0; float norm = 0.0f;
    if (lane == 0) {
        s = src[e]; d = dst[e];
        norm = dinv[s] * dinv[d];
    }
    s    = __shfl(s, 0);
    d    = __shfl(d, 0);
    norm = __shfl(norm, 0);

    const float* hp = h   + (size_t)s * D + lane * PER;
    float*       ap = agg + (size_t)d * D + lane * PER;

    float v[PER];
    if (PER == 4) {
        float4 x4 = *(const float4*)hp;
        v[0] = x4.x; v[1] = x4.y; v[2] = x4.z; v[3] = x4.w;
    } else {
        float2 x2 = *(const float2*)hp;
        v[0] = x2.x; v[1] = x2.y;
    }
#pragma unroll
    for (int j = 0; j < PER; ++j) atomicAdd(ap + j, norm * v[j]);
}

// ---------------- WMMA strip GEMM: C[N x M] = A[N x 128] @ W, K = 128 -----------
// One wave computes a full 16 x M strip: per K-chunk the A fragment is loaded
// once and reused across MT independent column-tile WMMAs (no D->A/B hazards,
// B loads overlap WMMA issue). ABF16 selects the A dtype at compile time so
// the inner loop has no control flow.
// A: row-major, row stride 128.  Bt: bf16 [m][k] (transposed weight).

template <bool ABF16, int MT>
__global__ __launch_bounds__(256) void wmma_gemm_strip(const void* __restrict__ Aptr,
                                                       const __bf16* __restrict__ Bt,
                                                       float* __restrict__ C,
                                                       int row_tiles) {
    const int M = MT * 16;
    int wave = blockIdx.x * 8 + (threadIdx.x >> 5);   // 8 waves / 256-thread block
    int lane = threadIdx.x & 31;
    if (wave >= row_tiles) return;                    // wave-uniform: EXEC stays full

    int r  = lane & 15;        // A row within tile / B col within tile / D col
    int kh = lane >> 4;        // k-half selector

    int row = wave * 16 + r;

    v8f acc[MT];
#pragma unroll
    for (int t = 0; t < MT; ++t) acc[t] = (v8f){};

#pragma unroll
    for (int kc = 0; kc < 4; ++kc) {
        // ---- A fragment: 16x32 bf16. lanes 0-15: k in {0..7, 16..23}; 16-31: +8
        int k0a = kc * 32 + kh * 8;
        int k0b = k0a + 16;
        v16bf a;
        if (ABF16) {
            const __bf16* arow = (const __bf16*)Aptr + (size_t)row * 128;
            v8bf lo = *(const v8bf*)(arow + k0a);
            v8bf hi = *(const v8bf*)(arow + k0b);
#pragma unroll
            for (int j = 0; j < 8; ++j) { a[j] = lo[j]; a[j + 8] = hi[j]; }
        } else {
            const float* arow = (const float*)Aptr + (size_t)row * 128;
            float4 f0 = *(const float4*)(arow + k0a);
            float4 f1 = *(const float4*)(arow + k0a + 4);
            float4 f2 = *(const float4*)(arow + k0b);
            float4 f3 = *(const float4*)(arow + k0b + 4);
            a[0]  = (__bf16)f0.x; a[1]  = (__bf16)f0.y; a[2]  = (__bf16)f0.z; a[3]  = (__bf16)f0.w;
            a[4]  = (__bf16)f1.x; a[5]  = (__bf16)f1.y; a[6]  = (__bf16)f1.z; a[7]  = (__bf16)f1.w;
            a[8]  = (__bf16)f2.x; a[9]  = (__bf16)f2.y; a[10] = (__bf16)f2.z; a[11] = (__bf16)f2.w;
            a[12] = (__bf16)f3.x; a[13] = (__bf16)f3.y; a[14] = (__bf16)f3.z; a[15] = (__bf16)f3.w;
        }

        // ---- B fragment per column tile: 32x16 bf16.
        //      lanes 0-15: k in {0..7, 16..23}; 16-31: +8 (same as A).
        int kb0 = kc * 32 + kh * 8;
#pragma unroll
        for (int t = 0; t < MT; ++t) {
            const __bf16* bcol = Bt + (size_t)(t * 16 + r) * 128;
            v8bf blo = *(const v8bf*)(bcol + kb0);
            v8bf bhi = *(const v8bf*)(bcol + kb0 + 16);
            v16bf b;
#pragma unroll
            for (int j = 0; j < 8; ++j) { b[j] = blo[j]; b[j + 8] = bhi[j]; }

            acc[t] = __builtin_amdgcn_wmma_f32_16x16x32_bf16(
                         /*neg_a=*/false, a, /*neg_b=*/false, b,
                         /*c_mod=*/(short)0, acc[t], /*reuse_a=*/false, /*reuse_b=*/false);
        }
    }

    // ---- D store: lane&15 = col n; vgpr v holds m = (lane>>4)*8 + v
    int n  = lane & 15;
    int mb = (lane >> 4) * 8;
#pragma unroll
    for (int t = 0; t < MT; ++t) {
#pragma unroll
        for (int v = 0; v < 8; ++v) {
            C[(size_t)(wave * 16 + mb + v) * M + t * 16 + n] = acc[t][v];
        }
    }
}

// ---------------------------------------------------------------------------

extern "C" void kernel_launch(void* const* d_in, const int* in_sizes, int n_in,
                              void* d_out, int out_size, void* d_ws, size_t ws_size,
                              hipStream_t stream) {
    const float* x  = (const float*)d_in[0];
    const int*   ei = (const int*)d_in[1];   // [2, E]; src = ei, dst = ei + E
    const float* W1 = (const float*)d_in[2]; // [128,128]
    const float* b1 = (const float*)d_in[3]; // [128]
    const float* W2 = (const float*)d_in[4]; // [128,64]
    const float* b2 = (const float*)d_in[5]; // [64]
    float* out = (float*)d_out;              // [N,64]

    const int N = NN, K = 128, H = 128, O = 64;
    const int E = in_sizes[1] / 2;

    // ---- workspace carve-out (~103 MB), 256B-aligned chunks
    char* ws = (char*)d_ws;
    size_t off = 0;
    auto carve = [&](size_t bytes) {
        void* p = ws + off;
        off = (off + bytes + 255) & ~(size_t)255;
        return p;
    };
    float*  dinv = (float*)carve((size_t)N * 4);
    __bf16* w1t  = (__bf16*)carve((size_t)H * K * 2);
    __bf16* w2t  = (__bf16*)carve((size_t)O * H * 2);
    float*  h1   = (float*)carve((size_t)N * H * 4);  // GEMM1 out; later reused as hbf
    float*  agg1 = (float*)carve((size_t)N * H * 4);  // layer-1 agg; later reused as h2
    __bf16* hbf  = (__bf16*)h1;   // relu(agg1+b1) in bf16 (fits in h1's space)
    float*  h2   = agg1;          // GEMM2 out (agg1 dead by then)

    const int TB = 256;
    const int rtiles = N / 16;                   // 6250, exact
    const int gemm_blocks = (rtiles + 7) / 8;    // 8 waves (strips) per block

    // ---- degree -> dinv
    k_fill_ones<<<(N + TB - 1) / TB, TB, 0, stream>>>(dinv, N);
    k_count_deg<<<(E + TB - 1) / TB, TB, 0, stream>>>(ei + E, dinv, E);
    k_rsqrt<<<(N + TB - 1) / TB, TB, 0, stream>>>(dinv, N);

    // ---- weights -> bf16, transposed
    k_w_transpose_bf16<<<(K * H + TB - 1) / TB, TB, 0, stream>>>(W1, w1t, K, H);
    k_w_transpose_bf16<<<(H * O + TB - 1) / TB, TB, 0, stream>>>(W2, w2t, H, O);

    // ---- layer 1: h1 = x @ W1 (WMMA) ; agg = D^-1/2 A D^-1/2 h1 ; relu(+b1) -> bf16
    {
        wmma_gemm_strip<false, 8><<<gemm_blocks, TB, 0, stream>>>(x, w1t, h1, rtiles);
        int nd = N * H;
        k_selfloop_init<<<(nd + TB - 1) / TB, TB, 0, stream>>>(h1, dinv, nullptr, agg1, N, H);
        unsigned sblocks = ((unsigned)E * 32u + TB - 1) / TB;
        k_scatter<4><<<sblocks, TB, 0, stream>>>(ei, ei + E, dinv, h1, agg1, E);
        k_bias_relu_bf16<<<(nd + TB - 1) / TB, TB, 0, stream>>>(agg1, b1, hbf, N, H);
    }

    // ---- layer 2: h2 = hbf @ W2 (WMMA) ; out = D^-1/2 A D^-1/2 h2 + b2
    {
        wmma_gemm_strip<true, 4><<<gemm_blocks, TB, 0, stream>>>(hbf, w2t, h2, rtiles);
        int nd = N * O;
        k_selfloop_init<<<(nd + TB - 1) / TB, TB, 0, stream>>>(h2, dinv, b2, out, N, O);
        unsigned sblocks = ((unsigned)E * 32u + TB - 1) / TB;
        k_scatter<2><<<sblocks, TB, 0, stream>>>(ei, ei + E, dinv, h2, out, E);
    }
}
